// MultiHeadAttention_22316650070631
// MI455X (gfx1250) — matrix operand order repaired and checked by reading, not compiled, not run on hardware
//
#include <hip/hip_runtime.h>
#include <hip/hip_bf16.h>

typedef __attribute__((ext_vector_type(16))) _Float16 v16h;
typedef __attribute__((ext_vector_type(8)))  _Float16 v8h;
typedef __attribute__((ext_vector_type(8)))  float    v8f;
typedef __attribute__((ext_vector_type(4)))  int      v4i;

static constexpr int Bv = 4, Sv = 2048, Ev = 1024, Hv = 16, Dv = 64;

// ---- optional gfx1250 async global->LDS path (guarded; falls back cleanly) ----
#if defined(__has_builtin)
#if __has_builtin(__builtin_amdgcn_global_load_async_to_lds_b128) && \
    __has_builtin(__builtin_amdgcn_s_wait_asynccnt)
#define USE_ASYNC_COPY 1
#endif
#endif
#ifndef USE_ASYNC_COPY
#define USE_ASYNC_COPY 0
#endif

// copy 16 bytes (8 x f16) from global to LDS, per lane
__device__ __forceinline__ void copy16_g2l(const _Float16* g, _Float16* l) {
#if USE_ASYNC_COPY
  __builtin_amdgcn_global_load_async_to_lds_b128(
      (__attribute__((address_space(1))) v4i*)(v4i*)(void*)g,
      (__attribute__((address_space(3))) v4i*)(v4i*)(void*)l, 0, 0);
#else
  *(v8h*)l = *(const v8h*)g;
#endif
}

__device__ __forceinline__ void async_copy_wait() {
#if USE_ASYNC_COPY
  __builtin_amdgcn_s_wait_asynccnt(0);
#endif
}

__device__ __forceinline__ int lane_id() { return threadIdx.x & 31; }

// ---- WMMA fragment loaders (contiguous k = 16*(lane/16) + j in A and B, wave32) ----

// A fragment: 16x32 (MxK) f16 from row-major source with leading dim ld.
__device__ __forceinline__ v16h load_a_frag(const _Float16* base, int ld) {
  const int lane = lane_id();
  const int row  = lane & 15;
  const int kh   = (lane >> 4) << 4;            // 0 or 16
  v16h a;
#pragma unroll
  for (int j = 0; j < 16; ++j) {
    const int k = kh + j;
    a[j] = base[row * ld + k];
  }
  return a;
}

// B fragment: 32x16 (KxN), source stored as [N][K] row-major (value(k,n)=base[n*ld+k])
__device__ __forceinline__ v16h load_b_frag_nk(const _Float16* base, int ld) {
  const int lane = lane_id();
  const int n  = lane & 15;
  const int kh = (lane >> 4) << 4;              // 0 or 16
  v16h b;
#pragma unroll
  for (int j = 0; j < 16; ++j) b[j] = base[n * ld + kh + j];
  return b;
}

// B fragment: 32x16 (KxN), source stored as [K][N] row-major
__device__ __forceinline__ v16h load_b_frag_kn(const _Float16* base, int ld) {
  const int lane = lane_id();
  const int n  = lane & 15;
  const int kh = (lane >> 4) << 4;
  v16h b;
#pragma unroll
  for (int j = 0; j < 16; ++j) b[j] = base[(kh + j) * ld + n];
  return b;
}

__device__ __forceinline__ v8f wmma16(v16h a, v16h b, v8f c) {
  return __builtin_amdgcn_wmma_f32_16x16x32_f16(false, a, false, b, (short)0, c,
                                                false, false);
}

// =====================================================================
// Kernel 1: per-head QKV projections  out[bh,s,d] = sum_e x[b,s,e]*W[h,d,e]
// grid = (S/128, B*H, 3), block = 256 (8 waves); 128x64 tile, K-chunks of 32
// =====================================================================
__global__ __launch_bounds__(256) void qkv_proj_kernel(
    const float* __restrict__ Qin, const float* __restrict__ Kin,
    const float* __restrict__ Vin, const float* __restrict__ Wq,
    const float* __restrict__ Wk, const float* __restrict__ Wv,
    _Float16* __restrict__ qkv) {
  constexpr int LD = 40;                         // 32 + pad
  __shared__ _Float16 Xs[128 * LD];
  __shared__ _Float16 Ws[64 * LD];

  const int z = blockIdx.z;
  const float* X = (z == 0) ? Qin : (z == 1) ? Kin : Vin;
  const float* W = (z == 0) ? Wq  : (z == 1) ? Wk  : Wv;
  _Float16* outp = qkv + (size_t)z * Bv * Hv * Sv * Dv;

  const int bh = blockIdx.y;
  const int b = bh / Hv, h = bh % Hv;
  const int s0 = blockIdx.x * 128;
  const int t = threadIdx.x;
  const int wave = t >> 5, lane = t & 31;
  const int rw = wave * 16;
  const int half = lane >> 4, col = lane & 15;

  const float* Xbase = X + ((size_t)b * Sv + s0) * Ev;
  const float* Wbase = W + (size_t)h * Dv * Ev;

  v8f acc[4] = {};

  for (int ko = 0; ko < Ev; ko += 32) {
    // stage X tile 128x32 f32 -> f16 (1024 float4)
#pragma unroll
    for (int i = 0; i < 4; ++i) {
      const int id = t + i * 256;
      const int r = id >> 3, c4 = (id & 7) << 2;
      const float4 xv = *(const float4*)(Xbase + (size_t)r * Ev + ko + c4);
      _Float16* d = &Xs[r * LD + c4];
      d[0] = (_Float16)xv.x; d[1] = (_Float16)xv.y;
      d[2] = (_Float16)xv.z; d[3] = (_Float16)xv.w;
    }
    // stage W tile 64x32 (512 float4): Ws[n][k] = W[h, n, ko+k]
#pragma unroll
    for (int i = 0; i < 2; ++i) {
      const int id = t + i * 256;
      const int r = id >> 3, c4 = (id & 7) << 2;
      const float4 wv = *(const float4*)(Wbase + (size_t)r * Ev + ko + c4);
      _Float16* d = &Ws[r * LD + c4];
      d[0] = (_Float16)wv.x; d[1] = (_Float16)wv.y;
      d[2] = (_Float16)wv.z; d[3] = (_Float16)wv.w;
    }
    if (ko + 32 < Ev)
      __builtin_prefetch(Xbase + (size_t)(t >> 1) * Ev + ko + 32 + (t & 1) * 16, 0, 1);
    __syncthreads();

    const v16h a = load_a_frag(&Xs[rw * LD], LD);
#pragma unroll
    for (int n = 0; n < 4; ++n)
      acc[n] = wmma16(a, load_b_frag_nk(&Ws[(n * 16) * LD], LD), acc[n]);
    __syncthreads();
  }

#pragma unroll
  for (int n = 0; n < 4; ++n)
#pragma unroll
    for (int i = 0; i < 8; ++i) {
      const int s = s0 + rw + i + 8 * half;
      outp[((size_t)bh * Sv + s) * Dv + n * 16 + col] = (_Float16)acc[n][i];
    }
}

// =====================================================================
// Kernel 2: flash attention per (b,h).  grid = (S/128, B*H), block = 256.
// Each wave owns a 16-row query strip; 64-key chunks with online softmax.
// =====================================================================
__global__ __launch_bounds__(256) void attn_kernel(
    const _Float16* __restrict__ qb, const _Float16* __restrict__ kb,
    const _Float16* __restrict__ vb, const int* __restrict__ mask,
    _Float16* __restrict__ att) {
  constexpr int LD = 72;                          // 64 + pad
  __shared__ _Float16 Qs[128 * LD];
  __shared__ _Float16 Kc[64 * LD];
  __shared__ _Float16 Vc[64 * LD];
  __shared__ _Float16 Ps[8 * 16 * LD];            // per-wave P staging (C->A relayout)

  const int bh = blockIdx.y;
  const int b = bh / Hv, h = bh % Hv;
  const int s0 = blockIdx.x * 128;
  const int t = threadIdx.x, wave = t >> 5, lane = t & 31;
  const int rw = wave * 16;
  const int half = lane >> 4, col = lane & 15;

  // load Q strip 128x64 f16 (1024 x 16B, 4 per thread) -- async if available
  const _Float16* qsrc = qb + ((size_t)bh * Sv + s0) * Dv;
#pragma unroll
  for (int i = 0; i < 4; ++i) {
    const int id = t + i * 256;
    const int r = id >> 3, c8 = (id & 7) << 3;
    copy16_g2l(qsrc + (size_t)r * Dv + c8, &Qs[r * LD + c8]);
  }

  v8f o[4] = {};
  float mrow[8], lrow[8];
#pragma unroll
  for (int i = 0; i < 8; ++i) { mrow[i] = -__builtin_huge_valf(); lrow[i] = 0.f; }

  const _Float16* ksrc = kb + (size_t)bh * Sv * Dv;
  const _Float16* vsrc = vb + (size_t)bh * Sv * Dv;
  const int* mbase = mask + (size_t)b * Sv * Sv;

  for (int n0 = 0; n0 < Sv; n0 += 64) {
    __syncthreads();                               // Kc/Vc safe to overwrite
#pragma unroll
    for (int i = 0; i < 2; ++i) {
      const int id = t + i * 256;
      const int r = id >> 3, c8 = (id & 7) << 3;
      copy16_g2l(ksrc + (size_t)(n0 + r) * Dv + c8, &Kc[r * LD + c8]);
      copy16_g2l(vsrc + (size_t)(n0 + r) * Dv + c8, &Vc[r * LD + c8]);
    }
    async_copy_wait();                             // this wave's async writes done
    __syncthreads();                               // all waves' tiles visible

    // S = Q K^T  (16x64 per wave); K-dim = D = 64 -> two WMMA K-steps
    v8f sc[4] = {};
#pragma unroll
    for (int kk = 0; kk < 64; kk += 32) {
      const v16h aq = load_a_frag(&Qs[rw * LD + kk], LD);
#pragma unroll
      for (int ct = 0; ct < 4; ++ct)
        sc[ct] = wmma16(aq, load_b_frag_nk(&Kc[(ct * 16) * LD + kk], LD), sc[ct]);
    }
    // scale + mask
#pragma unroll
    for (int ct = 0; ct < 4; ++ct)
#pragma unroll
      for (int i = 0; i < 8; ++i) {
        const int q = s0 + rw + i + 8 * half;
        const int kc = n0 + ct * 16 + col;
        float s = sc[ct][i] * 0.125f;              // 1/sqrt(64)
        if (mbase[(size_t)q * Sv + kc] == 0) s = -__builtin_huge_valf();
        sc[ct][i] = s;
      }
    // online softmax update (rows live across 16 lanes of each half-wave)
#pragma unroll
    for (int i = 0; i < 8; ++i) {
      float cm = sc[0][i];
#pragma unroll
      for (int ct = 1; ct < 4; ++ct) cm = fmaxf(cm, sc[ct][i]);
#pragma unroll
      for (int off = 1; off < 16; off <<= 1) cm = fmaxf(cm, __shfl_xor(cm, off, 32));
      const float nm = fmaxf(mrow[i], cm);
      const float corr = __expf(mrow[i] - nm);
      mrow[i] = nm;
      lrow[i] *= corr;
#pragma unroll
      for (int ct = 0; ct < 4; ++ct) o[ct][i] *= corr;
      float rs = 0.f;
#pragma unroll
      for (int ct = 0; ct < 4; ++ct) {
        const float p = __expf(sc[ct][i] - nm);
        sc[ct][i] = p;
        rs += p;
      }
#pragma unroll
      for (int off = 1; off < 16; off <<= 1) rs += __shfl_xor(rs, off, 32);
      lrow[i] += rs;
    }
    // P (C-layout) -> LDS -> A-layout for second GEMM
    _Float16* pw = &Ps[wave * 16 * LD];
#pragma unroll
    for (int ct = 0; ct < 4; ++ct)
#pragma unroll
      for (int i = 0; i < 8; ++i)
        pw[(i + 8 * half) * LD + ct * 16 + col] = (_Float16)sc[ct][i];
    __syncthreads();

    // O += P V   (K-dim = 64 keys -> two WMMA K-steps)
#pragma unroll
    for (int kk = 0; kk < 64; kk += 32) {
      const v16h ap = load_a_frag(&Ps[wave * 16 * LD + kk], LD);
#pragma unroll
      for (int ct = 0; ct < 4; ++ct)
        o[ct] = wmma16(ap, load_b_frag_kn(&Vc[kk * LD + ct * 16], LD), o[ct]);
    }
  }

  // normalize and store concat-head layout [B*S, H*D] (f16)
#pragma unroll
  for (int i = 0; i < 8; ++i) {
    const int q = s0 + rw + i + 8 * half;
    const float inv = 1.0f / lrow[i];
#pragma unroll
    for (int ct = 0; ct < 4; ++ct)
      att[(size_t)(b * Sv + q) * (Hv * Dv) + h * Dv + ct * 16 + col] =
          (_Float16)(o[ct][i] * inv);
  }
}

// =====================================================================
// Kernel 3: out = att[8192x1024] @ Wo^T + bo.  grid = (B*S/128, E/64)
// =====================================================================
__global__ __launch_bounds__(256) void out_proj_kernel(
    const _Float16* __restrict__ att, const float* __restrict__ Wo,
    const float* __restrict__ bo, float* __restrict__ out) {
  constexpr int LD = 40;
  constexpr int Kd = Hv * Dv;                     // 1024
  __shared__ _Float16 As[128 * LD];
  __shared__ _Float16 Bs[64 * LD];

  const int r0 = blockIdx.x * 128;                // token rows
  const int n0 = blockIdx.y * 64;                 // output features
  const int t = threadIdx.x, wave = t >> 5, lane = t & 31;
  const int rw = wave * 16;
  const int half = lane >> 4, col = lane & 15;

  v8f acc[4] = {};
  for (int ko = 0; ko < Kd; ko += 32) {
    // A tile 128x32 f16 (512 x 16B) -- async if available
#pragma unroll
    for (int i = 0; i < 2; ++i) {
      const int id = t + i * 256;
      const int r = id >> 2, c8 = (id & 3) << 3;
      copy16_g2l(att + (size_t)(r0 + r) * Kd + ko + c8, &As[r * LD + c8]);
    }
    // B tile: Bs[n][k] = Wo[n0+n, ko+k] (512 float4, f32->f16)
#pragma unroll
    for (int i = 0; i < 2; ++i) {
      const int id = t + i * 256;
      const int r = id >> 3, c4 = (id & 7) << 2;
      const float4 wv = *(const float4*)(Wo + (size_t)(n0 + r) * Kd + ko + c4);
      _Float16* d = &Bs[r * LD + c4];
      d[0] = (_Float16)wv.x; d[1] = (_Float16)wv.y;
      d[2] = (_Float16)wv.z; d[3] = (_Float16)wv.w;
    }
    if (ko + 32 < Kd)
      __builtin_prefetch(att + (size_t)(r0 + (t >> 1)) * Kd + ko + 32, 0, 1);
    async_copy_wait();
    __syncthreads();

    const v16h a = load_a_frag(&As[rw * LD], LD);
#pragma unroll
    for (int n = 0; n < 4; ++n)
      acc[n] = wmma16(a, load_b_frag_nk(&Bs[(n * 16) * LD], LD), acc[n]);
    __syncthreads();
  }

#pragma unroll
  for (int n = 0; n < 4; ++n) {
    const int e = n0 + n * 16 + col;
    const float bias = bo[e];
#pragma unroll
    for (int i = 0; i < 8; ++i) {
      const int r = r0 + rw + i + 8 * half;
      out[(size_t)r * Ev + e] = acc[n][i] + bias;
    }
  }
}

// =====================================================================
extern "C" void kernel_launch(void* const* d_in, const int* in_sizes, int n_in,
                              void* d_out, int out_size, void* d_ws, size_t ws_size,
                              hipStream_t stream) {
  (void)in_sizes; (void)n_in; (void)out_size; (void)ws_size;
  const float* values  = (const float*)d_in[0];
  const float* keys    = (const float*)d_in[1];
  const float* queries = (const float*)d_in[2];
  const int*   mask    = (const int*)d_in[3];
  const float* Wq = (const float*)d_in[4];
  const float* Wk = (const float*)d_in[5];
  const float* Wv = (const float*)d_in[6];
  const float* Wo = (const float*)d_in[7];
  const float* bo = (const float*)d_in[8];
  float* out = (float*)d_out;

  const size_t perT = (size_t)Bv * Hv * Sv * Dv;  // 8,388,608 elems per tensor
  _Float16* qkv = (_Float16*)d_ws;                // [3][B*H][S][D] f16
  _Float16* att = qkv + 3 * perT;                 // [B*S][H*D] f16

  dim3 blk(256);
  qkv_proj_kernel<<<dim3(Sv / 128, Bv * Hv, 3), blk, 0, stream>>>(
      queries, keys, values, Wq, Wk, Wv, qkv);
  attn_kernel<<<dim3(Sv / 128, Bv * Hv), blk, 0, stream>>>(
      qkv, qkv + perT, qkv + 2 * perT, mask, att);
  out_proj_kernel<<<dim3((Bv * Sv) / 128, Ev / 64), blk, 0, stream>>>(
      att, Wo, bo, out);
}
